// CWRRTCell_28080496181590
// MI455X (gfx1250) — matrix operand order repaired and checked by reading, not compiled, not run on hardware
//
#include <hip/hip_runtime.h>

// ---------------------------------------------------------------------------
// CWRRT cell for MI455X (gfx1250): bf16 WMMA GEMMs + fused epilogues.
// B=8192, D=512, H=8, MEM=16, HD=64, DFF=2048.
// GEMM: block tile 128x128, BK=32, 8 waves (4x2), wave tile 32x64
//       (2x4 WMMA frags), LDS double-buffered (1 barrier / K-step).
// ---------------------------------------------------------------------------

#define CB    8192
#define CD    512
#define CMEM  16
#define CDFF  2048

typedef __attribute__((ext_vector_type(16))) __bf16 v16bf;
typedef __attribute__((ext_vector_type(8)))  float  v8f;
typedef __attribute__((ext_vector_type(2)))  __bf16 bf2;

__device__ __forceinline__ float sigmoidf_(float v) {
    return 1.0f / (1.0f + __expf(-v));
}
__device__ __forceinline__ float gelu_tanh_(float v) {
    float c = 0.7978845608028654f * (v + 0.044715f * v * v * v);
    return 0.5f * v * (1.0f + tanhf(c));
}

// ---------------------------------------------------------------------------
// Kernel 1: x_in = x + ssum * sigmoid(alpha)
// ---------------------------------------------------------------------------
__global__ __launch_bounds__(256) void prep_xin_kernel(
    const float* __restrict__ x, const float* __restrict__ ssum,
    const float* __restrict__ alpha, float* __restrict__ xin) {
    size_t i = (size_t)blockIdx.x * 256 + threadIdx.x;   // over B*D
    int d = (int)(i & (CD - 1));
    xin[i] = x[i] + ssum[i] * sigmoidf_(alpha[d]);
}

// ---------------------------------------------------------------------------
// Kernel 2: updated_mem = concat(mem[:,1:,:], x_in)   (float4 granularity)
// ---------------------------------------------------------------------------
__global__ __launch_bounds__(256) void shift_mem_kernel(
    const float4* __restrict__ mem, const float4* __restrict__ xin,
    float4* __restrict__ umem) {
    size_t i = (size_t)blockIdx.x * 256 + threadIdx.x;   // over B*MEM*(D/4)
    int c = (int)(i & 127);          // float4 within row (D/4 = 128)
    size_t bs = i >> 7;              // b*MEM + s
    int s = (int)(bs & (CMEM - 1));
    size_t b = bs >> 4;
    umem[i] = (s < CMEM - 1) ? mem[i + 128] : xin[b * 128 + c];
}

// ---------------------------------------------------------------------------
// Kernel 3: row LayerNorm (one block per row of 512)
// ---------------------------------------------------------------------------
__global__ __launch_bounds__(256) void ln_kernel(
    const float* __restrict__ x, const float* __restrict__ scale,
    const float* __restrict__ bias, float* __restrict__ y) {
    int b = blockIdx.x, t = threadIdx.x;
    const float* xr = x + (size_t)b * CD;
    float2 v = *(const float2*)(xr + t * 2);
    float s = v.x + v.y;
    float ss = v.x * v.x + v.y * v.y;
    #pragma unroll
    for (int off = 16; off; off >>= 1) {
        s  += __shfl_xor(s,  off, 32);
        ss += __shfl_xor(ss, off, 32);
    }
    __shared__ float sb[8], sqb[8];
    int wave = t >> 5, lane = t & 31;
    if (lane == 0) { sb[wave] = s; sqb[wave] = ss; }
    __syncthreads();
    if (t == 0) {
        float S = 0.f, SS = 0.f;
        #pragma unroll
        for (int i = 0; i < 8; ++i) { S += sb[i]; SS += sqb[i]; }
        sb[0] = S; sqb[0] = SS;
    }
    __syncthreads();
    float mu  = sb[0] * (1.0f / CD);
    float var = sqb[0] * (1.0f / CD) - mu * mu;
    float r   = rsqrtf(var + 1e-6f);
    int d = t * 2;
    float* yr = y + (size_t)b * CD;
    yr[d]     = (v.x - mu) * r * scale[d]     + bias[d];
    yr[d + 1] = (v.y - mu) * r * scale[d + 1] + bias[d + 1];
}

// ---------------------------------------------------------------------------
// WMMA GEMM: C[M,N] = act(A[M,K] @ W[K,N] + bias [+ res]) ; bf16 tensor ops.
// MODE 0: fp32 out          MODE 1: bf16 out (K/V)
// MODE 2: fp32 out + res    MODE 3: gelu
// MODE 4: xo = acc+b+res -> out; out2 = ssum*sig(lam) + xo*(1-sig(lam))
// ---------------------------------------------------------------------------
#define BMT 128
#define BNT 128
#define BKT 32
#define LDAS 40   // bf16 units per LDS row, 80B stride (16B aligned)
#define LDBS 40

template <int MODE>
__global__ __launch_bounds__(256) void gemm_wmma_kernel(
    const float* __restrict__ A, int M, int Kd, int lda,
    const float* __restrict__ W, int ldw,
    const float* __restrict__ bias,
    const float* __restrict__ res,
    const float* __restrict__ aux_ssum,
    const float* __restrict__ aux_lam,
    float* __restrict__ out, float* __restrict__ out2,
    __bf16* __restrict__ outbf, int ldc) {
    __shared__ __align__(16) __bf16 As[2][BMT * LDAS];
    __shared__ __align__(16) __bf16 Bs[2][BNT * LDBS];

    const int tid  = threadIdx.x;
    const int wave = tid >> 5, lane = tid & 31;
    const int wm = wave >> 1, wn = wave & 1;      // 4x2 wave grid
    const int l16 = lane & 15, half = lane >> 4;
    const int rowA0 = blockIdx.x * BMT;
    const int colB0 = blockIdx.y * BNT;

    // staging indices
    const int ar = tid >> 1;               // A row 0..127
    const int ac = (tid & 1) * 16;         // A col 0 or 16
    const int bk = tid >> 3;               // B row (k) 0..31
    const int bc = (tid & 7) * 16;         // B col base (16 consecutive cols)

    v8f acc[2][4] = {};
    float4 ra[4], rb[4];

    const int nk = Kd / BKT;

    // ---- prologue: stage tile 0 ----
    {
        const float* srcA = A + (size_t)(rowA0 + ar) * lda + ac;
        #pragma unroll
        for (int i = 0; i < 4; ++i) ra[i] = *(const float4*)(srcA + i * 4);
        const float* srcB = W + (size_t)bk * ldw + colB0 + bc;
        #pragma unroll
        for (int i = 0; i < 4; ++i) rb[i] = *(const float4*)(srcB + i * 4);
    }
    {
        __bf16* dA = &As[0][ar * LDAS + ac];
        #pragma unroll
        for (int i = 0; i < 4; ++i) {
            dA[i * 4 + 0] = (__bf16)ra[i].x; dA[i * 4 + 1] = (__bf16)ra[i].y;
            dA[i * 4 + 2] = (__bf16)ra[i].z; dA[i * 4 + 3] = (__bf16)ra[i].w;
        }
        #pragma unroll
        for (int i = 0; i < 4; ++i) {
            Bs[0][(bc + i * 4 + 0) * LDBS + bk] = (__bf16)rb[i].x;
            Bs[0][(bc + i * 4 + 1) * LDBS + bk] = (__bf16)rb[i].y;
            Bs[0][(bc + i * 4 + 2) * LDBS + bk] = (__bf16)rb[i].z;
            Bs[0][(bc + i * 4 + 3) * LDBS + bk] = (__bf16)rb[i].w;
        }
    }
    __syncthreads();

    for (int t = 0; t < nk; ++t) {
        const int cur = t & 1;
        const bool more = (t + 1) < nk;
        // ---- issue global loads for tile t+1 (hidden under WMMAs) ----
        if (more) {
            int k0 = (t + 1) * BKT;
            const float* srcA = A + (size_t)(rowA0 + ar) * lda + k0 + ac;
            if (t + 2 < nk) __builtin_prefetch(srcA + BKT, 0, 3);
            #pragma unroll
            for (int i = 0; i < 4; ++i) ra[i] = *(const float4*)(srcA + i * 4);
            const float* srcB = W + (size_t)(k0 + bk) * ldw + colB0 + bc;
            #pragma unroll
            for (int i = 0; i < 4; ++i) rb[i] = *(const float4*)(srcB + i * 4);
        }
        // ---- fragments from LDS (ISA bf16 layouts) ----
        v16bf afr[2], bfr[4];
        #pragma unroll
        for (int i = 0; i < 2; ++i) {
            // A 16x32: half 0 -> K {0..7,16..23}; half 1 -> K {8..15,24..31}
            const __bf16* p = &As[cur][(wm * 32 + i * 16 + l16) * LDAS + half * 8];
            union { uint4 u[2]; v16bf v; } cv;
            cv.u[0] = *(const uint4*)p;
            cv.u[1] = *(const uint4*)(p + 16);
            afr[i] = cv.v;
        }
        #pragma unroll
        for (int j = 0; j < 4; ++j) {
            // B 32x16: half 0 -> K {0..7,16..23}; half 1 -> K {8..15,24..31}
            const __bf16* p = &Bs[cur][(wn * 64 + j * 16 + l16) * LDBS + half * 8];
            union { uint4 u[2]; v16bf v; } cv;
            cv.u[0] = *(const uint4*)p;
            cv.u[1] = *(const uint4*)(p + 16);
            bfr[j] = cv.v;
        }
        #pragma unroll
        for (int i = 0; i < 2; ++i)
            #pragma unroll
            for (int j = 0; j < 4; ++j)
                acc[i][j] = __builtin_amdgcn_wmma_f32_16x16x32_bf16(
                    false, afr[i], false, bfr[j], (short)0, acc[i][j],
                    false, false);
        // ---- store staged tile t+1 into the other buffer ----
        if (more) {
            __bf16* dA = &As[cur ^ 1][ar * LDAS + ac];
            #pragma unroll
            for (int i = 0; i < 4; ++i) {
                dA[i * 4 + 0] = (__bf16)ra[i].x; dA[i * 4 + 1] = (__bf16)ra[i].y;
                dA[i * 4 + 2] = (__bf16)ra[i].z; dA[i * 4 + 3] = (__bf16)ra[i].w;
            }
            #pragma unroll
            for (int i = 0; i < 4; ++i) {
                Bs[cur ^ 1][(bc + i * 4 + 0) * LDBS + bk] = (__bf16)rb[i].x;
                Bs[cur ^ 1][(bc + i * 4 + 1) * LDBS + bk] = (__bf16)rb[i].y;
                Bs[cur ^ 1][(bc + i * 4 + 2) * LDBS + bk] = (__bf16)rb[i].z;
                Bs[cur ^ 1][(bc + i * 4 + 3) * LDBS + bk] = (__bf16)rb[i].w;
            }
        }
        __syncthreads();
    }

    // ---- epilogue: C 16x16 fp32 layout: N = l16, M = r + 8*half ----
    #pragma unroll
    for (int i = 0; i < 2; ++i) {
        #pragma unroll
        for (int j = 0; j < 4; ++j) {
            int gcol = colB0 + wn * 64 + j * 16 + l16;
            float bv = bias[gcol];
            float lam_s = (MODE == 4) ? sigmoidf_(aux_lam[gcol]) : 0.f;
            #pragma unroll
            for (int r = 0; r < 8; ++r) {
                int grow = rowA0 + wm * 32 + i * 16 + half * 8 + r;
                size_t idx = (size_t)grow * ldc + gcol;
                float v = acc[i][j][r] + bv;
                if (MODE == 0) {
                    out[idx] = v;
                } else if (MODE == 1) {
                    outbf[idx] = (__bf16)v;
                } else if (MODE == 2) {
                    out[idx] = v + res[idx];
                } else if (MODE == 3) {
                    out[idx] = gelu_tanh_(v);
                } else { // MODE 4
                    float xo = v + res[idx];
                    out[idx]  = xo;
                    out2[idx] = aux_ssum[idx] * lam_s + xo * (1.0f - lam_s);
                }
            }
        }
    }
}

// ---------------------------------------------------------------------------
// Fused attention, block per batch row, wave per head.
// q fp32 [B,512]; K,V bf16 [B,16,512]; ctx fp32 [B,512].
// ---------------------------------------------------------------------------
__global__ __launch_bounds__(256) void attn_kernel(
    const float* __restrict__ q, const __bf16* __restrict__ Kb,
    const __bf16* __restrict__ Vb, float* __restrict__ ctx) {
    int b = blockIdx.x;
    int h = threadIdx.x >> 5;            // head = wave
    int lane = threadIdx.x & 31;
    int d0 = h * 64 + lane * 2;
    const __bf16* kb = Kb + (size_t)b * CMEM * CD;
    const __bf16* vb = Vb + (size_t)b * CMEM * CD;
    const float* qr = q + (size_t)b * CD;
    float q0 = qr[d0], q1 = qr[d0 + 1];

    float sc[CMEM];
    #pragma unroll
    for (int m = 0; m < CMEM; ++m) {
        bf2 kv = *(const bf2*)(kb + (size_t)m * CD + d0);
        float p = q0 * (float)kv[0] + q1 * (float)kv[1];
        #pragma unroll
        for (int off = 16; off; off >>= 1) p += __shfl_xor(p, off, 32);
        sc[m] = p * 0.125f;              // 1/sqrt(64)
    }
    float mx = sc[0];
    #pragma unroll
    for (int m = 1; m < CMEM; ++m) mx = fmaxf(mx, sc[m]);
    float sum = 0.f;
    #pragma unroll
    for (int m = 0; m < CMEM; ++m) { sc[m] = __expf(sc[m] - mx); sum += sc[m]; }
    float inv = 1.0f / sum;
    float c0 = 0.f, c1 = 0.f;
    #pragma unroll
    for (int m = 0; m < CMEM; ++m) {
        bf2 vv = *(const bf2*)(vb + (size_t)m * CD + d0);
        c0 += sc[m] * (float)vv[0];
        c1 += sc[m] * (float)vv[1];
    }
    float* cr = ctx + (size_t)b * CD;
    cr[d0] = c0 * inv;
    cr[d0 + 1] = c1 * inv;
}

// ---------------------------------------------------------------------------
// Launch
// ---------------------------------------------------------------------------
extern "C" void kernel_launch(void* const* d_in, const int* in_sizes, int n_in,
                              void* d_out, int out_size, void* d_ws, size_t ws_size,
                              hipStream_t stream) {
    const float* mem   = (const float*)d_in[0];
    const float* ssum  = (const float*)d_in[1];
    const float* x     = (const float*)d_in[2];
    const float* alpha = (const float*)d_in[3];
    const float* lam   = (const float*)d_in[4];
    const float* ln1s  = (const float*)d_in[5];
    const float* ln1b  = (const float*)d_in[6];
    const float* ln2s  = (const float*)d_in[7];
    const float* ln2b  = (const float*)d_in[8];
    const float* wq = (const float*)d_in[9];   const float* bq = (const float*)d_in[10];
    const float* wk = (const float*)d_in[11];  const float* bk = (const float*)d_in[12];
    const float* wv = (const float*)d_in[13];  const float* bv = (const float*)d_in[14];
    const float* wo = (const float*)d_in[15];  const float* bo = (const float*)d_in[16];
    const float* w1 = (const float*)d_in[17];  const float* b1 = (const float*)d_in[18];
    const float* w2 = (const float*)d_in[19];  const float* b2 = (const float*)d_in[20];

    float* outF       = (float*)d_out;
    float* umem       = outF;                                   // [B,16,D]
    float* new_ssum   = outF + (size_t)CB * CMEM * CD;          // [B,D]
    float* x_out      = new_ssum + (size_t)CB * CD;             // [B,D]

    const size_t BD = (size_t)CB * CD;
    float* ws   = (float*)d_ws;
    float* xin  = ws;                 // [B,D]
    float* y1   = ws + 1 * BD;        // [B,D]
    float* qv   = ws + 2 * BD;        // [B,D]
    float* ctx  = ws + 3 * BD;        // [B,D]
    float* xmid = ws + 4 * BD;        // [B,D]
    float* y2   = ws + 5 * BD;        // [B,D]
    float* h1   = ws + 6 * BD;        // [B,4D] -> 4*BD floats
    __bf16* Kb  = (__bf16*)(ws + 10 * BD);   // [B,16,D] bf16 -> 8*BD floats
    __bf16* Vb  = (__bf16*)(ws + 18 * BD);   // [B,16,D] bf16

    // 1. gated input
    prep_xin_kernel<<<(unsigned)(BD / 256), 256, 0, stream>>>(x, ssum, alpha, xin);
    // 2. sliding-window memory shift -> d_out.updated_mem
    shift_mem_kernel<<<(unsigned)((size_t)CB * CMEM * (CD / 4) / 256), 256, 0, stream>>>(
        (const float4*)mem, (const float4*)xin, (float4*)umem);
    // 3. LN1
    ln_kernel<<<CB, 256, 0, stream>>>(xin, ln1s, ln1b, y1);
    // 4-5. K,V projections over updated_mem (bf16 outputs)
    {
        dim3 g(CB * CMEM / BMT, CD / BNT);
        gemm_wmma_kernel<1><<<g, 256, 0, stream>>>(umem, CB * CMEM, CD, CD,
            wk, CD, bk, nullptr, nullptr, nullptr, nullptr, nullptr, Kb, CD);
        gemm_wmma_kernel<1><<<g, 256, 0, stream>>>(umem, CB * CMEM, CD, CD,
            wv, CD, bv, nullptr, nullptr, nullptr, nullptr, nullptr, Vb, CD);
    }
    // 6. q projection
    {
        dim3 g(CB / BMT, CD / BNT);
        gemm_wmma_kernel<0><<<g, 256, 0, stream>>>(y1, CB, CD, CD,
            wq, CD, bq, nullptr, nullptr, nullptr, qv, nullptr, nullptr, CD);
    }
    // 7. attention
    attn_kernel<<<CB, 256, 0, stream>>>(qv, Kb, Vb, ctx);
    // 8. output projection + residual -> x_mid
    {
        dim3 g(CB / BMT, CD / BNT);
        gemm_wmma_kernel<2><<<g, 256, 0, stream>>>(ctx, CB, CD, CD,
            wo, CD, bo, xin, nullptr, nullptr, xmid, nullptr, nullptr, CD);
    }
    // 9. LN2
    ln_kernel<<<CB, 256, 0, stream>>>(xmid, ln2s, ln2b, y2);
    // 10. MLP up + GELU
    {
        dim3 g(CB / BMT, CDFF / BNT);
        gemm_wmma_kernel<3><<<g, 256, 0, stream>>>(y2, CB, CD, CD,
            w1, CDFF, b1, nullptr, nullptr, nullptr, h1, nullptr, nullptr, CDFF);
    }
    // 11. MLP down + residual + gated state update -> x_out, new_ssum
    {
        dim3 g(CB / BMT, CD / BNT);
        gemm_wmma_kernel<4><<<g, 256, 0, stream>>>(h1, CB, CDFF, CDFF,
            w2, CD, b2, xmid, ssum, lam, x_out, new_ssum, nullptr, CD);
    }
}
